// Decoder_Curriculum_39264591020279
// MI455X (gfx1250) — matrix operand order repaired and checked by reading, not compiled, not run on hardware
//
#include <hip/hip_runtime.h>
#include <hip/hip_bf16.h>

typedef __attribute__((ext_vector_type(16))) _Float16 v16h;
typedef __attribute__((ext_vector_type(8)))  float    v8f;

#define BATCH 16384
#define EMB   64
#define HDIM  128
#define MLPD  1024
#define GSC   1024
#define TLEN  14
#define ROWS  16      // persons per scene
#define BN_EPS 1e-5f

// ---------------- workspace layout (bytes) ----------------
#define ALIGN256(x) (((x) + 255) & ~((size_t)255))
// f16 weights
#define OFF_WCAT   ((size_t)0)                                   // 512 x 192
#define SZ_WCAT    ((size_t)512 * 192 * 2)
#define OFF_WH2P   ALIGN256(OFF_WCAT + SZ_WCAT)                  // 64 x 128
#define SZ_WH2P    ((size_t)64 * 128 * 2)
#define OFF_WPRE   ALIGN256(OFF_WH2P + SZ_WH2P)                  // 1024 x 128
#define SZ_WPRE    ((size_t)1024 * 128 * 2)
#define OFF_WPOOL  ALIGN256(OFF_WPRE + SZ_WPRE)                  // 128 x 1024
#define SZ_WPOOL   ((size_t)128 * 1024 * 2)
#define OFF_WM1    ALIGN256(OFF_WPOOL + SZ_WPOOL)                // 1024 x 256
#define SZ_WM1     ((size_t)1024 * 256 * 2)
#define OFF_WM2    ALIGN256(OFF_WM1 + SZ_WM1)                    // 128 x 1024
#define SZ_WM2     ((size_t)128 * 1024 * 2)
// activations
#define OFF_H16    ALIGN256(OFF_WM2 + SZ_WM2)                    // B x 128 f16
#define SZ_H16     ((size_t)BATCH * HDIM * 2)
#define OFF_CWS    ALIGN256(OFF_H16 + SZ_H16)                    // B x 128 f32
#define SZ_CWS     ((size_t)BATCH * HDIM * 4)
#define OFF_SEGMX  ALIGN256(OFF_CWS + SZ_CWS)                    // G x 128 f32
#define SZ_SEGMX   ((size_t)GSC * HDIM * 4)
#define OFF_SUM    ALIGN256(OFF_SEGMX + SZ_SEGMX)                // 128 f32
#define OFF_SUMSQ  ALIGN256(OFF_SUM + 512)                       // 128 f32

__device__ __forceinline__ float sigf(float x)   { return 1.0f / (1.0f + __expf(-x)); }
__device__ __forceinline__ float leakyf(float x) { return x > 0.0f ? x : 0.01f * x; }

// One wave computes a 16x16 tile of  A(16xK) * W(16 rows, K cols)^T, f32 accumulate.
// A: row-major with leading dim lda (halves). W: row-major (n,k) with leading dim ldw.
__device__ __forceinline__ void wmma_tile(const _Float16* __restrict__ A, int lda,
                                          const _Float16* __restrict__ W, int ldw,
                                          int K, v8f& acc, int lane) {
  const int  m  = lane & 15;
  const bool hi = lane >= 16;
  const _Float16* arow = A + m * lda + (hi ? 16 : 0);
  const _Float16* wrow = W + m * ldw + (hi ? 16 : 0);
  for (int kc = 0; kc < K; kc += 32) {
    v16h a, b;
#pragma unroll
    for (int j = 0; j < 8; ++j)  a[j]     = arow[kc + j];
#pragma unroll
    for (int j = 0; j < 8; ++j)  a[8 + j] = arow[kc + 8 + j];
#pragma unroll
    for (int j = 0; j < 16; ++j) b[j]     = wrow[kc + j];
    acc = __builtin_amdgcn_wmma_f32_16x16x32_f16(false, a, false, b, (short)0, acc,
                                                 false, false);
  }
}

// ---------------- init: convert weights to f16, seed h16/c ----------------
__global__ void __launch_bounds__(256) decgan_init(
    const float* __restrict__ Wih, const float* __restrict__ Whh,
    const float* __restrict__ Wh2p, const float* __restrict__ Wpre,
    const float* __restrict__ Wpool, const float* __restrict__ Wm1,
    const float* __restrict__ Wm2, const float* __restrict__ h0,
    const float* __restrict__ c0, _Float16* __restrict__ wcat,
    _Float16* __restrict__ wh2p, _Float16* __restrict__ wpre,
    _Float16* __restrict__ wpool, _Float16* __restrict__ wm1,
    _Float16* __restrict__ wm2, _Float16* __restrict__ h16,
    float* __restrict__ cws) {
  const long tid = (long)blockIdx.x * blockDim.x + threadIdx.x;
  const long stride = (long)gridDim.x * blockDim.x;
  for (long i = tid; i < 512L * 192; i += stride) {
    int n = (int)(i / 192), k = (int)(i % 192);
    float v = (k < 64) ? Wih[n * 64 + k] : Whh[n * 128 + (k - 64)];
    wcat[i] = (_Float16)v;
  }
  for (long i = tid; i < 64L * 128; i += stride)   wh2p[i]  = (_Float16)Wh2p[i];
  for (long i = tid; i < 1024L * 128; i += stride) wpre[i]  = (_Float16)Wpre[i];
  for (long i = tid; i < 128L * 1024; i += stride) wpool[i] = (_Float16)Wpool[i];
  for (long i = tid; i < 1024L * 256; i += stride) wm1[i]   = (_Float16)Wm1[i];
  for (long i = tid; i < 128L * 1024; i += stride) wm2[i]   = (_Float16)Wm2[i];
  for (long i = tid; i < (long)BATCH * HDIM; i += stride) {
    h16[i] = (_Float16)h0[i];
    cws[i] = c0[i];
  }
}

__global__ void decgan_zero_stats(float* __restrict__ sum, float* __restrict__ sumsq) {
  int t = threadIdx.x;
  if (t < HDIM) { sum[t] = 0.0f; sumsq[t] = 0.0f; }
}

// ---------------- K1: LSTM + hidden2pose + pre-pool MLP + pool + scene max ------
// LDS layout (bytes):
//   [0, 33280)      gates f32 16x(ld 520)   ALIAS  dh f16 16x(ld 1032)
//   [33280, 39680)  acat f16 16x(ld 200)    cols 0..63 = x, 64..191 = h
//   [39680, 47872)  cbuf f32 16x128
//   [47872, 56320)  ph   f32 16x(ld 132)
__global__ void __launch_bounds__(256) decgan_k1(
    const float* __restrict__ x_src,       // (B, 64) f32 (last_pos or prev pred slice)
    float* __restrict__ pred_out,          // (B, 64) f32  = d_out + t*B*64
    const _Float16* __restrict__ wcat,     // 512 x 192
    const _Float16* __restrict__ wh2p,     // 64 x 128
    const _Float16* __restrict__ wpre,     // 1024 x 128
    const _Float16* __restrict__ wpool,    // 128 x 1024
    const float* __restrict__ b_ih, const float* __restrict__ b_hh,
    const float* __restrict__ b_h2p, const float* __restrict__ b_pre,
    const float* __restrict__ b_pool,
    _Float16* __restrict__ h16,            // (B,128) f16, read old / write new
    float* __restrict__ cws,               // (B,128) f32
    float* __restrict__ segmax,            // (G,128) f32
    float* __restrict__ sum, float* __restrict__ sumsq) {
  __shared__ __align__(16) unsigned char smem[56320];
  float*    gates = (float*)smem;                    // ld 520
  _Float16* dh    = (_Float16*)smem;                 // ld 1032 (alias)
  _Float16* acat  = (_Float16*)(smem + 33280);       // ld 200
  float*    cbuf  = (float*)(smem + 39680);          // ld 128
  float*    ph    = (float*)(smem + 47872);          // ld 132

  const int s    = blockIdx.x;     // scene
  const int tid  = threadIdx.x;
  const int wave = tid >> 5;
  const int lane = tid & 31;
  const int m     = lane & 15;
  const int rbase = (lane >= 16) ? 8 : 0;

  // phase 1: stage x (f32->f16), h (f16), c (f32)
  for (int e = tid; e < ROWS * EMB; e += 256) {
    int r = e >> 6, j = e & 63;
    acat[r * 200 + j] = (_Float16)x_src[(size_t)(s * ROWS + r) * EMB + j];
  }
  for (int e = tid; e < ROWS * HDIM; e += 256) {
    int r = e >> 7, j = e & 127;
    acat[r * 200 + 64 + j] = h16[(size_t)(s * ROWS + r) * HDIM + j];
    cbuf[r * 128 + j]      = cws[(size_t)(s * ROWS + r) * HDIM + j];
  }
  __syncthreads();

  // phase 2: gates = [x|h] @ [W_ih|W_hh]^T  (16x512, K=192) -> 32 tiles
  for (int nt = wave; nt < 32; nt += 8) {
    const int n0 = nt * 16;
    v8f acc = {};
    wmma_tile(acat, 200, wcat + (size_t)n0 * 192, 192, 192, acc, lane);
#pragma unroll
    for (int r = 0; r < 8; ++r) gates[(r + rbase) * 520 + n0 + m] = acc[r];
  }
  __syncthreads();

  // phase 3: LSTM pointwise (f32)
  for (int e = tid; e < ROWS * HDIM; e += 256) {
    int r = e >> 7, j = e & 127;
    int row = s * ROWS + r;
    float gi = gates[r * 520 + j]       + b_ih[j]       + b_hh[j];
    float gf = gates[r * 520 + 128 + j] + b_ih[128 + j] + b_hh[128 + j];
    float gg = gates[r * 520 + 256 + j] + b_ih[256 + j] + b_hh[256 + j];
    float go = gates[r * 520 + 384 + j] + b_ih[384 + j] + b_hh[384 + j];
    float c  = sigf(gf) * cbuf[r * 128 + j] + sigf(gi) * tanhf(gg);
    float h  = sigf(go) * tanhf(c);
    cws[(size_t)row * HDIM + j] = c;
    _Float16 h16v = (_Float16)h;
    acat[r * 200 + 64 + j] = h16v;
    h16[(size_t)row * HDIM + j] = h16v;
  }
  __syncthreads();

  // phase 4a: real_pose = h @ W_h2p^T + b + x   (16x64, K=128) -> 4 tiles
  if (wave < 4) {
    const int n0 = wave * 16;
    v8f acc = {};
    wmma_tile(acat + 64, 200, wh2p + (size_t)n0 * 128, 128, 128, acc, lane);
    const int col = n0 + m;
#pragma unroll
    for (int r = 0; r < 8; ++r) {
      int row = s * ROWS + r + rbase;
      float v = acc[r] + b_h2p[col] + x_src[(size_t)row * EMB + col];
      pred_out[(size_t)row * EMB + col] = v;
    }
  }
  // phase 4b: dh = leaky(h @ W_pre^T + b)  (16x1024, K=128) -> 64 tiles
  for (int nt = wave; nt < 64; nt += 8) {
    const int n0 = nt * 16;
    v8f acc = {};
    wmma_tile(acat + 64, 200, wpre + (size_t)n0 * 128, 128, 128, acc, lane);
    const int col = n0 + m;
#pragma unroll
    for (int r = 0; r < 8; ++r)
      dh[(r + rbase) * 1032 + col] = (_Float16)leakyf(acc[r] + b_pre[col]);
  }
  __syncthreads();

  // phase 5: ph = dh @ W_pool^T + b  (16x128, K=1024) -> 8 tiles
  {
    const int n0 = wave * 16;
    v8f acc = {};
    wmma_tile(dh, 1032, wpool + (size_t)n0 * 1024, 1024, 1024, acc, lane);
    const int col = n0 + m;
#pragma unroll
    for (int r = 0; r < 8; ++r) ph[(r + rbase) * 132 + col] = acc[r] + b_pool[col];
  }
  __syncthreads();

  // phase 6: per-scene max + BN stats over scene rows (mean over B == mean over G)
  if (tid < HDIM) {
    float mx = ph[tid];
#pragma unroll
    for (int r = 1; r < ROWS; ++r) mx = fmaxf(mx, ph[r * 132 + tid]);
    segmax[(size_t)s * HDIM + tid] = mx;
    atomicAdd(&sum[tid], mx);
    atomicAdd(&sumsq[tid], mx * mx);
  }
}

// ---------------- K2: BN + concat MLP [2H->MLP->H] ----------------
// LDS: acat2 f16 16x(ld 264) [h | pool_norm]  8448B ; dh2 f16 16x(ld 1032) 33024B
__global__ void __launch_bounds__(256) decgan_k2(
    const _Float16* __restrict__ wm1,      // 1024 x 256
    const _Float16* __restrict__ wm2,      // 128 x 1024
    const float* __restrict__ b_m1, const float* __restrict__ b_m2,
    const float* __restrict__ bn_g, const float* __restrict__ bn_b,
    const float* __restrict__ segmax, const float* __restrict__ sum,
    const float* __restrict__ sumsq,
    _Float16* __restrict__ h16,            // (B,128) read old h, write new h
    float* __restrict__ hout) {            // final h f32 (last step) or nullptr
  __shared__ __align__(16) unsigned char smem[8448 + 33024];
  _Float16* acat2 = (_Float16*)smem;             // ld 264
  _Float16* dh2   = (_Float16*)(smem + 8448);    // ld 1032

  const int s    = blockIdx.x;
  const int tid  = threadIdx.x;
  const int wave = tid >> 5;
  const int lane = tid & 31;
  const int m     = lane & 15;
  const int rbase = (lane >= 16) ? 8 : 0;

  if (tid < HDIM) {
    const int j = tid;
    float mu  = sum[j] * (1.0f / (float)GSC);
    float var = sumsq[j] * (1.0f / (float)GSC) - mu * mu;
    float rs  = rsqrtf(var + BN_EPS);
    float pn  = (segmax[(size_t)s * HDIM + j] - mu) * rs * bn_g[j] + bn_b[j];
    _Float16 p = (_Float16)pn;
#pragma unroll
    for (int r = 0; r < ROWS; ++r) acat2[r * 264 + 128 + j] = p;
  }
  for (int e = tid; e < ROWS * HDIM; e += 256) {
    int r = e >> 7, j = e & 127;
    acat2[r * 264 + j] = h16[(size_t)(s * ROWS + r) * HDIM + j];
  }
  __syncthreads();

  // dh2 = leaky([h|pool] @ W_m1^T + b)  (16x1024, K=256)
  for (int nt = wave; nt < 64; nt += 8) {
    const int n0 = nt * 16;
    v8f acc = {};
    wmma_tile(acat2, 264, wm1 + (size_t)n0 * 256, 256, 256, acc, lane);
    const int col = n0 + m;
#pragma unroll
    for (int r = 0; r < 8; ++r)
      dh2[(r + rbase) * 1032 + col] = (_Float16)leakyf(acc[r] + b_m1[col]);
  }
  __syncthreads();

  // h_new = leaky(dh2 @ W_m2^T + b)  (16x128, K=1024)
  {
    const int n0 = wave * 16;
    v8f acc = {};
    wmma_tile(dh2, 1032, wm2 + (size_t)n0 * 1024, 1024, 1024, acc, lane);
    const int col = n0 + m;
#pragma unroll
    for (int r = 0; r < 8; ++r) {
      int row = s * ROWS + r + rbase;
      float v = leakyf(acc[r] + b_m2[col]);
      h16[(size_t)row * HDIM + col] = (_Float16)v;
      if (hout) hout[(size_t)row * HDIM + col] = v;
    }
  }
}

extern "C" void kernel_launch(void* const* d_in, const int* in_sizes, int n_in,
                              void* d_out, int out_size, void* d_ws, size_t ws_size,
                              hipStream_t stream) {
  const float* last_pos = (const float*)d_in[0];
  const float* h0    = (const float*)d_in[1];
  const float* c0    = (const float*)d_in[2];
  const float* W_ih  = (const float*)d_in[3];
  const float* W_hh  = (const float*)d_in[4];
  const float* b_ih  = (const float*)d_in[5];
  const float* b_hh  = (const float*)d_in[6];
  const float* W_h2p = (const float*)d_in[7];
  const float* b_h2p = (const float*)d_in[8];
  const float* W_pre = (const float*)d_in[9];
  const float* b_pre = (const float*)d_in[10];
  const float* W_pool = (const float*)d_in[11];
  const float* b_pool = (const float*)d_in[12];
  const float* bn_g  = (const float*)d_in[13];
  const float* bn_b  = (const float*)d_in[14];
  const float* W_m1  = (const float*)d_in[15];
  const float* b_m1  = (const float*)d_in[16];
  const float* W_m2  = (const float*)d_in[17];
  const float* b_m2  = (const float*)d_in[18];

  float* out = (float*)d_out;
  unsigned char* ws = (unsigned char*)d_ws;

  _Float16* wcat  = (_Float16*)(ws + OFF_WCAT);
  _Float16* wh2p  = (_Float16*)(ws + OFF_WH2P);
  _Float16* wpre  = (_Float16*)(ws + OFF_WPRE);
  _Float16* wpool = (_Float16*)(ws + OFF_WPOOL);
  _Float16* wm1   = (_Float16*)(ws + OFF_WM1);
  _Float16* wm2   = (_Float16*)(ws + OFF_WM2);
  _Float16* h16   = (_Float16*)(ws + OFF_H16);
  float*    cws   = (float*)(ws + OFF_CWS);
  float*    segmx = (float*)(ws + OFF_SEGMX);
  float*    sum   = (float*)(ws + OFF_SUM);
  float*    sumsq = (float*)(ws + OFF_SUMSQ);

  decgan_init<<<2048, 256, 0, stream>>>(W_ih, W_hh, W_h2p, W_pre, W_pool, W_m1, W_m2,
                                        h0, c0, wcat, wh2p, wpre, wpool, wm1, wm2,
                                        h16, cws);

  float* h_final = out + (size_t)TLEN * BATCH * EMB;  // (1,B,128) region
  for (int t = 0; t < TLEN; ++t) {
    decgan_zero_stats<<<1, 128, 0, stream>>>(sum, sumsq);
    const float* x_src = (t == 0) ? last_pos : out + (size_t)(t - 1) * BATCH * EMB;
    float* pred = out + (size_t)t * BATCH * EMB;
    decgan_k1<<<GSC, 256, 0, stream>>>(x_src, pred, wcat, wh2p, wpre, wpool,
                                       b_ih, b_hh, b_h2p, b_pre, b_pool,
                                       h16, cws, segmx, sum, sumsq);
    decgan_k2<<<GSC, 256, 0, stream>>>(wm1, wm2, b_m1, b_m2, bn_g, bn_b,
                                       segmx, sum, sumsq, h16,
                                       (t == TLEN - 1) ? h_final : nullptr);
  }
}
